// FAENet_79096117723895
// MI455X (gfx1250) — matrix operand order repaired and checked by reading, not compiled, not run on hardware
//
#include <hip/hip_runtime.h>

// ---------------------------------------------------------------------------
// CDNA5 (gfx1250) FAENet forward. wave32, v_wmma_f32_16x16x32_f16 everywhere.
// All GEMM operands pre-converted to f16 (weights once per launch, activations
// written as f16 shadows by producer epilogues) so the WMMA inner loop is pure
// 16B loads + WMMA. Each wave computes a 16x32 tile (A fragment reused, 2 WMMA
// per K-step). Message-passing fuses filter-GEMM + gather + scatter-add.
// ---------------------------------------------------------------------------

typedef __attribute__((ext_vector_type(16))) _Float16 v16h;
typedef __attribute__((ext_vector_type(8)))  _Float16 v8h;
typedef __attribute__((ext_vector_type(8)))  float    v8f;
typedef __attribute__((ext_vector_type(4)))  float    v4f;

__device__ __forceinline__ float silu(float x) {
    return x / (1.0f + __expf(-x));
}

// Load 16 f16 A-elements per lane, contiguous K order as the callers pass it:
// elems 0..7 -> K = half*16 + i ; elems 8..15 -> K = half*16 + 8 + i
__device__ __forceinline__ v16h load_frag16(const _Float16* p0, const _Float16* p1) {
    v8h lo = *(const v8h*)p0;
    v8h hi = *(const v8h*)p1;
    v16h a;
#pragma unroll
    for (int i = 0; i < 8; ++i) { a[i] = lo[i]; a[8 + i] = hi[i]; }
    return a;
}

// ---- One 16x32 output tile (two accumulators): C = A[16,K] @ W^T + bias ----
// A: row-major [M,K] f16. W: row-major [Ncols,K] f16 (dense => x @ w.T).
__device__ __forceinline__ void wmma_tile2_core(const _Float16* __restrict__ A,
                                                const _Float16* __restrict__ W,
                                                const float* __restrict__ bias,
                                                int M0, int N0, int K, int M,
                                                v8f& c0, v8f& c1, int& m_base, int& n0) {
    const int lane = threadIdx.x & 31;
    const int r    = lane & 15;
    const int half = lane >> 4;
    n0     = N0 + r;                   // second column set is n0 + 16
    m_base = M0 + (half << 3);

    int rowA = M0 + r;
    if (rowA > M - 1) rowA = M - 1;    // clamp loads; stores are guarded
    const _Float16* arow = A + (size_t)rowA * K;
    const _Float16* w0   = W + (size_t)n0 * K;
    const _Float16* w1   = W + (size_t)(n0 + 16) * K;

    const float b0 = bias[n0], b1 = bias[n0 + 16];
#pragma unroll
    for (int j = 0; j < 8; ++j) { c0[j] = b0; c1[j] = b1; }

#pragma unroll 4
    for (int k0 = 0; k0 < K; k0 += 32) {
        // A fragment (reused by both WMMAs)
        v16h a = load_frag16(arow + k0 + half * 16, arow + k0 + half * 16 + 8);
        // B fragments: lane column, K = half*16 + i (16 contiguous halves)
        v16h bf0 = load_frag16(w0 + k0 + half * 16, w0 + k0 + half * 16 + 8);
        v16h bf1 = load_frag16(w1 + k0 + half * 16, w1 + k0 + half * 16 + 8);
        c0 = __builtin_amdgcn_wmma_f32_16x16x32_f16(false, a, false, bf0,
                                                    (short)0, c0, false, false);
        c1 = __builtin_amdgcn_wmma_f32_16x16x32_f16(false, a, false, bf1,
                                                    (short)0, c1, false, false);
    }
}

__device__ __forceinline__ bool tile_coords2(int Ncols, int M, int& tM, int& tN) {
    int wid     = (int)(((long long)blockIdx.x * blockDim.x + threadIdx.x) >> 5);
    int tilesN2 = Ncols >> 5;          // 32 output columns per wave
    tM = wid / tilesN2;
    tN = wid - tM * tilesN2;
    return tM < ((M + 15) >> 4);
}

// ---- GEMM + bias + SiLU, f16 output --------------------------------------
__global__ void k_gemm2_silu_f16(const _Float16* __restrict__ A, const _Float16* __restrict__ W,
                                 const float* __restrict__ bias, _Float16* __restrict__ Out,
                                 int M, int K, int Ncols) {
    int tM, tN;
    if (!tile_coords2(Ncols, M, tM, tN)) return;
    v8f c0, c1; int mb, n0;
    wmma_tile2_core(A, W, bias, tM * 16, tN * 32, K, M, c0, c1, mb, n0);
#pragma unroll
    for (int j = 0; j < 8; ++j) {
        int m = mb + j;
        if (m < M) {
            Out[(size_t)m * Ncols + n0]      = (_Float16)silu(c0[j]);
            Out[(size_t)m * Ncols + n0 + 16] = (_Float16)silu(c1[j]);
        }
    }
}

// ---- GEMM + bias + SiLU, f32 output --------------------------------------
__global__ void k_gemm2_silu_f32(const _Float16* __restrict__ A, const _Float16* __restrict__ W,
                                 const float* __restrict__ bias, float* __restrict__ Out,
                                 int M, int K, int Ncols) {
    int tM, tN;
    if (!tile_coords2(Ncols, M, tM, tN)) return;
    v8f c0, c1; int mb, n0;
    wmma_tile2_core(A, W, bias, tM * 16, tN * 32, K, M, c0, c1, mb, n0);
#pragma unroll
    for (int j = 0; j < 8; ++j) {
        int m = mb + j;
        if (m < M) {
            Out[(size_t)m * Ncols + n0]      = silu(c0[j]);
            Out[(size_t)m * Ncols + n0 + 16] = silu(c1[j]);
        }
    }
}

// ---- GEMM + bias + SiLU, dual f32 + f16-shadow output --------------------
__global__ void k_gemm2_silu_dual(const _Float16* __restrict__ A, const _Float16* __restrict__ W,
                                  const float* __restrict__ bias, float* __restrict__ Out32,
                                  _Float16* __restrict__ Out16, int M, int K, int Ncols) {
    int tM, tN;
    if (!tile_coords2(Ncols, M, tM, tN)) return;
    v8f c0, c1; int mb, n0;
    wmma_tile2_core(A, W, bias, tM * 16, tN * 32, K, M, c0, c1, mb, n0);
#pragma unroll
    for (int j = 0; j < 8; ++j) {
        int m = mb + j;
        if (m < M) {
            float v0 = silu(c0[j]), v1 = silu(c1[j]);
            size_t o0 = (size_t)m * Ncols + n0, o1 = o0 + 16;
            Out32[o0] = v0;             Out32[o1] = v1;
            Out16[o0] = (_Float16)v0;   Out16[o1] = (_Float16)v1;
        }
    }
}

// ---- Fused: e_l = silu(e @ geom^T + b); agg[dst] += e_l * hd[src] --------
__global__ void k_gemm2_msg(const _Float16* __restrict__ A, const _Float16* __restrict__ W,
                            const float* __restrict__ bias,
                            const int* __restrict__ src, const int* __restrict__ dst,
                            const float* __restrict__ hd, float* __restrict__ agg,
                            int M, int K, int Ncols) {
    int tM, tN;
    if (!tile_coords2(Ncols, M, tM, tN)) return;
    v8f c0, c1; int mb, n0;
    wmma_tile2_core(A, W, bias, tM * 16, tN * 32, K, M, c0, c1, mb, n0);
#pragma unroll
    for (int j = 0; j < 8; ++j) {
        int e = mb + j;
        if (e < M) {
            int s = src[e], d = dst[e];
            const size_t sb = (size_t)s * Ncols, db = (size_t)d * Ncols;
            atomicAdd(&agg[db + n0],      silu(c0[j]) * hd[sb + n0]);
            atomicAdd(&agg[db + n0 + 16], silu(c1[j]) * hd[sb + n0 + 16]);
        }
    }
}

// ---- Fused: h32 += silu(x @ up^T + b); refresh f16 shadow of h -----------
__global__ void k_gemm2_up_res(const _Float16* __restrict__ A, const _Float16* __restrict__ W,
                               const float* __restrict__ bias, float* __restrict__ h32,
                               _Float16* __restrict__ h16, int M, int K, int Ncols) {
    int tM, tN;
    if (!tile_coords2(Ncols, M, tM, tN)) return;
    v8f c0, c1; int mb, n0;
    wmma_tile2_core(A, W, bias, tM * 16, tN * 32, K, M, c0, c1, mb, n0);
#pragma unroll
    for (int j = 0; j < 8; ++j) {
        int m = mb + j;
        if (m < M) {
            size_t o0 = (size_t)m * Ncols + n0, o1 = o0 + 16;
            float v0 = h32[o0] + silu(c0[j]);
            float v1 = h32[o1] + silu(c1[j]);
            h32[o0] = v0;             h32[o1] = v1;
            h16[o0] = (_Float16)v0;   h16[o1] = (_Float16)v1;
        }
    }
}

// ---- f32 -> f16 weight conversion (once per launch) ----------------------
__global__ void k_cvt_f16(const float* __restrict__ src, _Float16* __restrict__ dst,
                          long long n) {
    long long i = (long long)blockIdx.x * 256 + threadIdx.x;
    if (i < n) dst[i] = (_Float16)src[i];
}

// ---- Edge pre-embedding: e0 = silu([rel_pos@W1^T | edge_attr@W12^T] + b) --
__global__ void k_edge_embed(const float* __restrict__ rel_pos,
                             const float* __restrict__ edge_attr,
                             const float* __restrict__ w1, const float* __restrict__ b1,
                             const float* __restrict__ w12, const float* __restrict__ b12,
                             _Float16* __restrict__ e0, long long E) {
    __shared__ float sw1[64 * 3], sb1[64], sw12[64 * 50], sb12[64];
    for (int i = threadIdx.x; i < 64 * 3; i += 256) sw1[i] = w1[i];
    for (int i = threadIdx.x; i < 64; i += 256) { sb1[i] = b1[i]; sb12[i] = b12[i]; }
    for (int i = threadIdx.x; i < 64 * 50; i += 256) sw12[i] = w12[i];
    __syncthreads();
    long long idx = (long long)blockIdx.x * 256 + threadIdx.x;
    if (idx >= E * 128) return;
    long long e = idx >> 7;
    int f = (int)(idx & 127);
    float acc;
    if (f < 64) {
        const float* rp = rel_pos + e * 3;
        acc = sb1[f] + rp[0] * sw1[f * 3] + rp[1] * sw1[f * 3 + 1] + rp[2] * sw1[f * 3 + 2];
    } else {
        int g = f - 64;
        const float* ea = edge_attr + e * 50;
        acc = sb12[g];
#pragma unroll
        for (int k = 0; k < 50; ++k) acc += ea[k] * sw12[g * 50 + k];
    }
    e0[idx] = (_Float16)silu(acc);
}

// ---- Node embedding gather (f16, feeds WMMA A directly) -------------------
__global__ void k_node_gather(const int* __restrict__ z, const int* __restrict__ tag,
                              const float* __restrict__ emb, const float* __restrict__ temb,
                              _Float16* __restrict__ h0, long long N) {
    long long idx = (long long)blockIdx.x * 256 + threadIdx.x;
    if (idx >= N * 256) return;
    long long nr = idx >> 8;
    int c = (int)(idx & 255);
    float v = (c < 224) ? emb[(size_t)z[nr] * 224 + c]
                        : temb[(size_t)tag[nr] * 32 + (c - 224)];
    h0[idx] = (_Float16)v;
}

// ---- GraphNorm helpers ----------------------------------------------------
__global__ void k_zero(float* p, long long n) {
    long long i = (long long)blockIdx.x * 256 + threadIdx.x;
    if (i < n) p[i] = 0.0f;
}
__global__ void k_colsum(const float* __restrict__ a, float* __restrict__ sums, int N) {
    int col = threadIdx.x;                 // blockDim.x == 128
    float acc = 0.0f;
    for (int row = blockIdx.x; row < N; row += gridDim.x)
        acc += a[(size_t)row * 128 + col];
    atomicAdd(&sums[col], acc);
}
__global__ void k_colvar(const float* __restrict__ a, const float* __restrict__ sums,
                         const float* __restrict__ gms, float* __restrict__ ss, int N) {
    int col = threadIdx.x;
    const float mean = sums[col] / (float)N;
    const float ms = gms[col];
    float acc = 0.0f;
    for (int row = blockIdx.x; row < N; row += gridDim.x) {
        float cen = a[(size_t)row * 128 + col] - mean * ms;
        acc += cen * cen;
    }
    atomicAdd(&ss[col], acc);
}
// act(GraphNorm(agg)) -> f16 buffer (consumed only as WMMA A-operand)
__global__ void k_gn(const float* __restrict__ a, const float* __restrict__ sums,
                     const float* __restrict__ ss, const float* __restrict__ gms,
                     const float* __restrict__ gw, const float* __restrict__ gb,
                     _Float16* __restrict__ out16, int N) {
    long long idx = (long long)blockIdx.x * 256 + threadIdx.x;
    if (idx >= (long long)N * 128) return;
    int col = (int)(idx & 127);
    float mean = sums[col] / (float)N;
    float var  = ss[col] / (float)N;
    float cen  = a[idx] - mean * gms[col];
    float g    = gw[col] * cen * rsqrtf(var + 1e-5f) + gb[col];
    out16[idx] = (_Float16)silu(g);
}

// ---- Output head: node_e = h1 @ out2_w^T + b; out[batch[n]] += node_e -----
__global__ void k_outfin(const float* __restrict__ h1, const float* __restrict__ w2,
                         const float* __restrict__ b2, const int* __restrict__ batch,
                         float* __restrict__ out, int N) {
    int n = blockIdx.x * blockDim.x + threadIdx.x;
    if (n >= N) return;
    const v4f* hr = (const v4f*)(h1 + (size_t)n * 128);
    const v4f* wr = (const v4f*)w2;
    float acc = b2[0];
#pragma unroll
    for (int i = 0; i < 32; ++i) {
        v4f hv = hr[i], wv = wr[i];
        acc += hv[0] * wv[0] + hv[1] * wv[1] + hv[2] * wv[2] + hv[3] * wv[3];
    }
    atomicAdd(&out[batch[n]], acc);
}

// ---------------------------------------------------------------------------
extern "C" void kernel_launch(void* const* d_in, const int* in_sizes, int n_in,
                              void* d_out, int out_size, void* d_ws, size_t ws_size,
                              hipStream_t stream) {
    const long long N = in_sizes[0];          // nodes
    const long long E = in_sizes[2] / 3;      // edges
    const int H = 256, F = 128, L = 3;

    const int*   z        = (const int*)  d_in[0];
    const int*   tag      = (const int*)  d_in[1];
    const float* rel_pos  = (const float*)d_in[2];
    const float* edge_attr= (const float*)d_in[3];
    const int*   ei       = (const int*)  d_in[4];
    const int*   batch    = (const int*)  d_in[5];
    const float* lin_e1_w = (const float*)d_in[6];
    const float* lin_e1_b = (const float*)d_in[7];
    const float* lin_e12_w= (const float*)d_in[8];
    const float* lin_e12_b= (const float*)d_in[9];
    const float* lin_e2_w = (const float*)d_in[10];
    const float* lin_e2_b = (const float*)d_in[11];
    const float* emb_w    = (const float*)d_in[12];
    const float* tag_emb_w= (const float*)d_in[13];
    const float* lin_w    = (const float*)d_in[14];
    const float* lin_b    = (const float*)d_in[15];
    const float* lin2_w   = (const float*)d_in[16];
    const float* lin2_b   = (const float*)d_in[17];
    const float* geom_w   = (const float*)d_in[18];
    const float* geom_b   = (const float*)d_in[19];
    const float* down_w   = (const float*)d_in[20];
    const float* down_b   = (const float*)d_in[21];
    const float* up_w     = (const float*)d_in[22];
    const float* up_b     = (const float*)d_in[23];
    const float* gn_w     = (const float*)d_in[24];
    const float* gn_b     = (const float*)d_in[25];
    const float* gn_ms    = (const float*)d_in[26];
    const float* out1_w   = (const float*)d_in[27];
    const float* out1_b   = (const float*)d_in[28];
    const float* out2_w   = (const float*)d_in[29];
    const float* out2_b   = (const float*)d_in[30];

    const int* src = ei;
    const int* dst = ei + E;

    // ---- workspace carve-out ----
    size_t off = 0;
    auto carve = [&](size_t bytes) {
        void* p = (char*)d_ws + off;
        off += (bytes + 255) & ~(size_t)255;
        return p;
    };
    _Float16* e0h    = (_Float16*)carve((size_t)E * F * sizeof(_Float16));
    _Float16* eh     = (_Float16*)carve((size_t)E * F * sizeof(_Float16));
    float*    h32    = (float*)   carve((size_t)N * H * sizeof(float));
    _Float16* n16a   = (_Float16*)carve((size_t)N * H * sizeof(_Float16)); // f16 shadow of h
    _Float16* n16b   = (_Float16*)carve((size_t)N * H * sizeof(_Float16));
    float*    hd     = (float*)   carve((size_t)N * F * sizeof(float));    // also h1
    float*    agg    = (float*)   carve((size_t)N * F * sizeof(float));
    _Float16* gnact  = (_Float16*)carve((size_t)N * F * sizeof(_Float16));
    float*    sums   = (float*)   carve(128 * sizeof(float));
    float*    ss     = (float*)   carve(128 * sizeof(float));
    // f16 weight copies
    _Float16* w16_e2   = (_Float16*)carve((size_t)F * F * sizeof(_Float16));
    _Float16* w16_lin  = (_Float16*)carve((size_t)H * H * sizeof(_Float16));
    _Float16* w16_lin2 = (_Float16*)carve((size_t)H * H * sizeof(_Float16));
    _Float16* w16_geom = (_Float16*)carve((size_t)L * F * F * sizeof(_Float16));
    _Float16* w16_down = (_Float16*)carve((size_t)L * F * H * sizeof(_Float16));
    _Float16* w16_up   = (_Float16*)carve((size_t)L * H * F * sizeof(_Float16));
    _Float16* w16_out1 = (_Float16*)carve((size_t)F * H * sizeof(_Float16));
    (void)ws_size; (void)n_in;

    auto cvt = [&](const float* s, _Float16* d, long long n) {
        k_cvt_f16<<<(int)((n + 255) / 256), 256, 0, stream>>>(s, d, n);
    };
    auto gemm_grid2 = [](long long M, int Nc) {
        long long tiles = ((M + 15) / 16) * (Nc / 32);   // 1 wave per 16x32 tile
        return (int)((tiles + 7) / 8);                   // 8 waves / block
    };

    // 0) zero output + convert weights to f16 (tiny, once per launch)
    k_zero<<<1, 256, 0, stream>>>((float*)d_out, out_size);
    cvt(lin_e2_w, w16_e2,   (long long)F * F);
    cvt(lin_w,    w16_lin,  (long long)H * H);
    cvt(lin2_w,   w16_lin2, (long long)H * H);
    cvt(geom_w,   w16_geom, (long long)L * F * F);
    cvt(down_w,   w16_down, (long long)L * F * H);
    cvt(up_w,     w16_up,   (long long)L * H * F);
    cvt(out1_w,   w16_out1, (long long)F * H);

    // 1) edge pre-embedding -> e0 (f16)
    k_edge_embed<<<(int)((E * 128 + 255) / 256), 256, 0, stream>>>(
        rel_pos, edge_attr, lin_e1_w, lin_e1_b, lin_e12_w, lin_e12_b, e0h, E);

    // 2) e = silu(e0 @ lin_e2^T + b)  [E,128] (WMMA, f16 in/out)
    k_gemm2_silu_f16<<<gemm_grid2(E, F), 256, 0, stream>>>(
        e0h, w16_e2, lin_e2_b, eh, (int)E, F, F);

    // 3) node embedding: gather -> lin -> lin2 (WMMA)
    k_node_gather<<<(int)((N * 256 + 255) / 256), 256, 0, stream>>>(
        z, tag, emb_w, tag_emb_w, n16a, N);
    k_gemm2_silu_f16<<<gemm_grid2(N, H), 256, 0, stream>>>(
        n16a, w16_lin, lin_b, n16b, (int)N, H, H);
    k_gemm2_silu_dual<<<gemm_grid2(N, H), 256, 0, stream>>>(
        n16b, w16_lin2, lin2_b, h32, n16a, (int)N, H, H);

    // 4) interaction layers
    for (int l = 0; l < L; ++l) {
        const _Float16* gw = w16_geom + (size_t)l * F * F;
        const _Float16* dw = w16_down + (size_t)l * F * H;
        const _Float16* uw = w16_up   + (size_t)l * H * F;
        const float* gb  = geom_b + (size_t)l * F;
        const float* db  = down_b + (size_t)l * F;
        const float* ub  = up_b   + (size_t)l * H;
        const float* gnw = gn_w   + (size_t)l * F;
        const float* gnb = gn_b   + (size_t)l * F;
        const float* gms = gn_ms  + (size_t)l * F;

        k_zero<<<(int)((N * F + 255) / 256), 256, 0, stream>>>(agg, N * F);
        k_zero<<<1, 128, 0, stream>>>(sums, 128);
        k_zero<<<1, 128, 0, stream>>>(ss, 128);

        // hd = silu(h @ down^T + b)          (WMMA, K=256 -> 128, f32 out)
        k_gemm2_silu_f32<<<gemm_grid2(N, F), 256, 0, stream>>>(
            n16a, dw, db, hd, (int)N, H, F);

        // fused filter + message + scatter   (WMMA)
        k_gemm2_msg<<<gemm_grid2(E, F), 256, 0, stream>>>(
            eh, gw, gb, src, dst, hd, agg, (int)E, F, F);

        // GraphNorm (batch=None: global stats) + act -> f16
        k_colsum<<<1024, 128, 0, stream>>>(agg, sums, (int)N);
        k_colvar<<<1024, 128, 0, stream>>>(agg, sums, gms, ss, (int)N);
        k_gn<<<(int)((N * F + 255) / 256), 256, 0, stream>>>(
            agg, sums, ss, gms, gnw, gnb, gnact, (int)N);

        // h += silu(act(gn) @ up^T + b)      (WMMA, K=128 -> 256, fused residual)
        k_gemm2_up_res<<<gemm_grid2(N, H), 256, 0, stream>>>(
            gnact, uw, ub, h32, n16a, (int)N, F, H);
    }

    // 5) output head: h1 = silu(h @ out1^T + b) (WMMA), then per-graph sum
    k_gemm2_silu_f32<<<gemm_grid2(N, F), 256, 0, stream>>>(
        n16a, w16_out1, out1_b, hd, (int)N, H, F);
    k_outfin<<<(int)((N + 255) / 256), 256, 0, stream>>>(
        hd, out2_w, out2_b, batch, (float*)d_out, (int)N);
}
